// AttentionBlock_58566174049019
// MI455X (gfx1250) — compile-verified, matrix operand order checked by reading, not run on hardware
//
#include <hip/hip_runtime.h>
#include <math.h>

typedef __bf16 bf16_t;
typedef __attribute__((ext_vector_type(16))) __bf16 v16bf;
typedef __attribute__((ext_vector_type(8)))  float  v8f;

#define EMBED   256
#define SEQ     4096
#define NBATCH  4
#define NTOK    (SEQ * NBATCH)
#define QK_SCALE 0.0625f   // 1/sqrt(256)

// ---------------------------------------------------------------------------
// Fragment loaders for V_WMMA_F32_16X16X32_BF16.
// 16-bit A/B fragment layout (wave32): lane L holds row/col (L&15); element i
// maps to k = (i&7) + 8*(L>>4) + (i>=8 ? 16 : 0).  Elements 0..7 and 8..15 are
// each 8 contiguous bf16 (16 bytes) -> two b128 loads per fragment.
// ---------------------------------------------------------------------------
static __device__ inline v16bf load_frag_bf16(const bf16_t* __restrict__ base,
                                              int row, int stride, int k0, int lane) {
  const int half = (lane >> 4) & 1;
  const bf16_t* p = base + (size_t)row * stride + k0 + half * 8;
  union { v16bf v; uint4 q[2]; } u;
  u.q[0] = *(const uint4*)(p);
  u.q[1] = *(const uint4*)(p + 16);
  return u.v;
}

// Same layout, but source is fp32 (convert to bf16 on the fly).
static __device__ inline v16bf load_frag_cvt(const float* __restrict__ base,
                                             int row, int stride, int k0, int lane) {
  const int half = (lane >> 4) & 1;
  const float* p = base + (size_t)row * stride + k0 + half * 8;
  float4 f0 = *(const float4*)(p);
  float4 f1 = *(const float4*)(p + 4);
  float4 f2 = *(const float4*)(p + 16);
  float4 f3 = *(const float4*)(p + 20);
  float t[16] = {f0.x, f0.y, f0.z, f0.w, f1.x, f1.y, f1.z, f1.w,
                 f2.x, f2.y, f2.z, f2.w, f3.x, f3.y, f3.z, f3.w};
  union { v16bf v; bf16_t e[16]; } u;
#pragma unroll
  for (int i = 0; i < 16; ++i) u.e[i] = (bf16_t)t[i];
  return u.v;
}

static __device__ inline v8f wmma_bf16(v16bf a, v16bf b, v8f c) {
  // (neg_a, A, neg_b, B, c_mod, C, reuse_a, reuse_b)
  return __builtin_amdgcn_wmma_f32_16x16x32_bf16(false, a, false, b, (short)0, c,
                                                 false, false);
}

// ---------------------------------------------------------------------------
// Async staging: copy one wave's contiguous 16 KB K slice (32 token rows x
// 512 B) from global into LDS with GLOBAL_LOAD_ASYNC_TO_LDS_B128 (ASYNCcnt).
// The immediate offset is applied to BOTH the global and the LDS address
// (ISA 08_async_tensor §4.4), so one base address per lane suffices.
// ---------------------------------------------------------------------------
static __device__ inline void stage_k_async(bf16_t* ldsdst, const bf16_t* gsrc,
                                            int lane) {
  uint32_t    l = (uint32_t)(uintptr_t)ldsdst + lane * 16; // LDS offset (addr[31:0])
  const char* g = (const char*)gsrc + lane * 16;
#pragma unroll
  for (int i = 0; i < 32; ++i)
    asm volatile("global_load_async_to_lds_b128 %0, %1, off offset:%2"
                 :: "v"(l), "v"(g), "i"(i * 512) : "memory");
}

// ---------------------------------------------------------------------------
// Kernel 1: Q = (x @ Wq^T) * 1/sqrt(d), K = x @ Wk^T, V^T = (x @ Wv^T)^T
// One wave computes one 16x16 output tile; blockIdx.y selects which weight.
// ---------------------------------------------------------------------------
__global__ __launch_bounds__(256) void qkv_proj_kernel(
    const float* __restrict__ x, const float* __restrict__ Wq,
    const float* __restrict__ Wk, const float* __restrict__ Wv,
    bf16_t* __restrict__ Qb, bf16_t* __restrict__ Kb, bf16_t* __restrict__ Vt) {
  const int lane = threadIdx.x & 31;
  const int w    = threadIdx.x >> 5;
  const int t    = blockIdx.x * 8 + w;   // 0..16383 tiles
  const int rt   = t >> 4;               // token tile  0..1023
  const int ct   = t & 15;               // out-dim tile 0..15
  const int which = blockIdx.y;          // 0=Q 1=K 2=V
  const float* W = (which == 0) ? Wq : ((which == 1) ? Wk : Wv);

  v8f c = {};
#pragma unroll
  for (int kc = 0; kc < 8; ++kc) {       // K = 256 in 8 steps of 32
    v16bf a  = load_frag_cvt(x, rt * 16 + (lane & 15), EMBED, kc * 32, lane);
    v16bf bb = load_frag_cvt(W, ct * 16 + (lane & 15), EMBED, kc * 32, lane);
    c = wmma_bf16(a, bb, c);
  }

  const int half = (lane >> 4) & 1;
  const int n    = lane & 15;
  if (which == 2) {                       // V: store transposed [e][token]
#pragma unroll
    for (int v = 0; v < 8; ++v) {
      int tok = rt * 16 + v + 8 * half;
      int e   = ct * 16 + n;
      Vt[(size_t)e * NTOK + tok] = (bf16_t)c[v];
    }
  } else {                                // Q/K: row-major [token][e]
    bf16_t* dst = (which == 0) ? Qb : Kb;
    float   s   = (which == 0) ? QK_SCALE : 1.0f;
#pragma unroll
    for (int v = 0; v < 8; ++v) {
      int tok = rt * 16 + v + 8 * half;
      int e   = ct * 16 + n;
      dst[(size_t)tok * EMBED + e] = (bf16_t)(c[v] * s);
    }
  }
}

// ---------------------------------------------------------------------------
// Kernel 2: flash attention. Block = 4 waves = 16 queries; each wave owns a
// disjoint 32-key slice per iteration (split-K online softmax).  K slices are
// double-buffered in LDS via async copies; V streams directly from global as
// contiguous b128 loads (transposed layout).  Partials are merged across
// waves through LDS (ds_add_f32) at the end.
// ---------------------------------------------------------------------------
__global__ __launch_bounds__(128) void flash_attn_kernel(
    const bf16_t* __restrict__ Qb, const bf16_t* __restrict__ Kb,
    const bf16_t* __restrict__ Vt, float* __restrict__ out) {
  __shared__ __align__(16) bf16_t kbuf[2][4][32][EMBED]; // 128 KB, dbl-buffered
  __shared__ __align__(16) bf16_t pbuf[4][16][32];       // per-wave P restage
  __shared__ float red_m[4][16];
  __shared__ float red_l[4][16];
  __shared__ float red_L[16];
  __shared__ float oacc[16][EMBED];                      // 16 KB combine buffer

  const int tid  = threadIdx.x;
  const int lane = tid & 31;
  const int w    = tid >> 5;
  const int half = (lane >> 4) & 1;
  const int n    = lane & 15;
  const int b    = blockIdx.y;
  const int qflat0 = b * SEQ + blockIdx.x * 16;

  for (int i = tid; i < 16 * EMBED; i += 128) (&oacc[0][0])[i] = 0.0f;
  __syncthreads();

  // Prefetch first K tile (this wave's 32 keys of iteration 0).
  stage_k_async(&kbuf[0][w][0][0], Kb + (size_t)(b * SEQ + w * 32) * EMBED, lane);

  // Resident Q fragments (16 queries x 256 dims, bf16)
  v16bf qf[8];
#pragma unroll
  for (int kc = 0; kc < 8; ++kc)
    qf[kc] = load_frag_bf16(Qb, qflat0 + n, EMBED, kc * 32, lane);

  v8f o[16];
#pragma unroll
  for (int et = 0; et < 16; ++et) o[et] = (v8f){};
  float m8[8], l8[8];
#pragma unroll
  for (int v = 0; v < 8; ++v) { m8[v] = -INFINITY; l8[v] = 0.0f; }

  for (int it = 0; it < SEQ / 128; ++it) {
    const int key0  = it * 128 + w * 32;  // this wave's 32 keys
    const int kflat = b * SEQ + key0;

    // Prefetch next tile into the other buffer, then wait for the current one
    // (async loads complete in order: <=32 outstanding => current tile landed).
    if (it + 1 < SEQ / 128) {
      stage_k_async(&kbuf[(it + 1) & 1][w][0][0],
                    Kb + (size_t)(kflat + 128) * EMBED, lane);
      asm volatile("s_wait_asynccnt 0x20" ::: "memory");
    } else {
      asm volatile("s_wait_asynccnt 0x0" ::: "memory");
    }
    const bf16_t* kcur = &kbuf[it & 1][w][0][0];

    // S = Q @ K^T  (two 16x16 score tiles = 32 keys), K operand from LDS
    v8f c0 = {}, c1 = {};
#pragma unroll
    for (int kc = 0; kc < 8; ++kc) {
      v16bf k0f = load_frag_bf16(kcur, n,      EMBED, kc * 32, lane);
      v16bf k1f = load_frag_bf16(kcur, 16 + n, EMBED, kc * 32, lane);
      c0 = wmma_bf16(qf[kc], k0f, c0);
      c1 = wmma_bf16(qf[kc], k1f, c1);
    }

    // Online softmax: row = v + 8*half, row values live in lanes of one half.
    float alpha[8];
#pragma unroll
    for (int v = 0; v < 8; ++v) {
      float t = fmaxf(c0[v], c1[v]);
#pragma unroll
      for (int off = 1; off < 16; off <<= 1) t = fmaxf(t, __shfl_xor(t, off, 32));
      float mn = fmaxf(m8[v], t);
      alpha[v] = __expf(m8[v] - mn);
      float p0 = __expf(c0[v] - mn);
      float p1 = __expf(c1[v] - mn);
      c0[v] = p0; c1[v] = p1;
      float rs = p0 + p1;
#pragma unroll
      for (int off = 1; off < 16; off <<= 1) rs += __shfl_xor(rs, off, 32);
      l8[v] = l8[v] * alpha[v] + rs;
      m8[v] = mn;
    }
#pragma unroll
    for (int et = 0; et < 16; ++et)
#pragma unroll
      for (int v = 0; v < 8; ++v) o[et][v] *= alpha[v];

    // Restage P (C-layout f32) -> LDS bf16 row-major 16x32 for the A operand.
#pragma unroll
    for (int v = 0; v < 8; ++v) {
      int row = v + 8 * half;
      pbuf[w][row][n]      = (bf16_t)c0[v];
      pbuf[w][row][16 + n] = (bf16_t)c1[v];
    }
    asm volatile("s_wait_dscnt 0" ::: "memory");  // same-wave LDS RAW fence
    v16bf pa = load_frag_bf16(&pbuf[w][0][0], n, 32, 0, lane);

    // O += P @ V  (V^T gives contiguous b128 B-fragments)
#pragma unroll
    for (int et = 0; et < 16; ++et) {
      v16bf vf = load_frag_bf16(Vt, et * 16 + n, NTOK, kflat, lane);
      o[et] = wmma_bf16(pa, vf, o[et]);
    }
  }

  // Cross-wave split-K combine through LDS.
  if (n == 0) {
#pragma unroll
    for (int v = 0; v < 8; ++v) {
      red_m[w][v + 8 * half] = m8[v];
      red_l[w][v + 8 * half] = l8[v];
    }
  }
  __syncthreads();

  float scale8[8];
#pragma unroll
  for (int v = 0; v < 8; ++v) {
    int row = v + 8 * half;
    float M = red_m[0][row];
#pragma unroll
    for (int ww = 1; ww < 4; ++ww) M = fmaxf(M, red_m[ww][row]);
    float L = 0.0f;
#pragma unroll
    for (int ww = 0; ww < 4; ++ww) L += red_l[ww][row] * __expf(red_m[ww][row] - M);
    scale8[v] = __expf(m8[v] - M);
    if (w == 0 && n == 0) red_L[row] = L;
  }
#pragma unroll
  for (int et = 0; et < 16; ++et)
#pragma unroll
    for (int v = 0; v < 8; ++v)
      atomicAdd(&oacc[v + 8 * half][et * 16 + n], o[et][v] * scale8[v]);
  __syncthreads();

  for (int i = tid; i < 16 * EMBED; i += 128) {
    int row = i >> 8;
    int col = i & 255;
    out[(size_t)(qflat0 + row) * EMBED + col] = oacc[row][col] / red_L[row];
  }
}

// ---------------------------------------------------------------------------
extern "C" void kernel_launch(void* const* d_in, const int* in_sizes, int n_in,
                              void* d_out, int out_size, void* d_ws, size_t ws_size,
                              hipStream_t stream) {
  const float* x  = (const float*)d_in[0];
  const float* Wq = (const float*)d_in[1];
  const float* Wk = (const float*)d_in[2];
  const float* Wv = (const float*)d_in[3];

  bf16_t* Qb = (bf16_t*)d_ws;                     //  8 MB
  bf16_t* Kb = Qb + (size_t)NTOK * EMBED;         //  8 MB
  bf16_t* Vt = Kb + (size_t)NTOK * EMBED;         //  8 MB (transposed [E][NTOK])
  float*  out = (float*)d_out;

  // 1024 token-tiles x 16 col-tiles, 8 waves/block, grid.y selects Q/K/V.
  qkv_proj_kernel<<<dim3(2048, 3), 256, 0, stream>>>(x, Wq, Wk, Wv, Qb, Kb, Vt);
  // One block per 16 queries per batch.
  flash_attn_kernel<<<dim3(SEQ / 16, NBATCH), 128, 0, stream>>>(Qb, Kb, Vt, out);
}
